// DifferentiablePreisachModel_84018150244461
// MI455X (gfx1250) — matrix operand order repaired and checked by reading, not compiled, not run on hardware
//
#include <hip/hip_runtime.h>
#include <hip/hip_bf16.h>

// -----------------------------------------------------------------------------
// Differentiable Preisach model for MI455X (gfx1250, wave32).
//
// One lane = one mesh point (alpha, beta, density). The sequential 2048-step
// hysteresis scan lives in registers. Every 16 steps each wave owns a 16x32
// (time x meshpoint) state tile: converted to f16, staged in wave-private LDS,
// transposed into the V_WMMA_F32_16X16X32_F16 A-operand layout, and multiplied
// against B = density replicated across all 16 N-columns (f32 accumulate).
// Column N=0 of C holds the 16 partial dot products sum_k state[t][k]*w[k];
// lanes 0 and 16 atomically add them into a global per-timestep accumulator.
// Finalize applies /sum(density) + h. Problem is tiny (<250 KB traffic) and
// VALU/latency bound; WMMA replaces 16 cross-lane reductions per chunk.
// -----------------------------------------------------------------------------

typedef __attribute__((ext_vector_type(16))) _Float16 v16h;
typedef __attribute__((ext_vector_type(8)))  float    v8f;

#define TEMP_INV   1000.0f   // 1/TEMP, TEMP = 0.001
#define M_SCALE    1.0f
#define OFFSET_C   0.0f
#define H_SLOPE    1.0f

#define WAVES_PER_BLOCK 4
#define BLOCK_THREADS   (WAVES_PER_BLOCK * 32)
#define TILE_STRIDE     34   // halves; 17-dword row step spreads LDS banks

// 2*sigmoid(x)-1 with x already scaled by 1/TEMP. Saturates cleanly:
// exp(-x)->0 => +1 ; exp(-x)->inf => rcp(inf)=0 => -1.
__device__ __forceinline__ float sig2m1(float x) {
  float e = __expf(-x);
  return 2.0f * __builtin_amdgcn_rcpf(1.0f + e) - 1.0f;
}

__global__ void __launch_bounds__(256) zero_ws_kernel(float* ws, int n) {
  int i = blockIdx.x * blockDim.x + threadIdx.x;
  if (i < n) ws[i] = 0.0f;
}

__global__ void __launch_bounds__(256) density_sum_kernel(
    const float* __restrict__ density, float* __restrict__ sum_slot, int K) {
  __shared__ float sm[256];
  float s = 0.0f;
  for (int i = threadIdx.x; i < K; i += 256) s += density[i];
  sm[threadIdx.x] = s;
  __syncthreads();
  for (int off = 128; off > 0; off >>= 1) {
    if ((int)threadIdx.x < off) sm[threadIdx.x] += sm[threadIdx.x + off];
    __syncthreads();
  }
  if (threadIdx.x == 0) *sum_slot = sm[0];
}

// One 16-timestep chunk: scan -> f16 tile -> WMMA transpose -> matvec -> atomics.
// GUARD=false is the uniform fast path (whole chunk in range): no per-step or
// per-atomic bound checks, single exec-mask region around the 8 atomics.
template <bool GUARD>
__device__ __forceinline__ void preisach_chunk(
    int t0, int N, const float* __restrict__ h,
    float alpha, float beta, float& state, float& field,
    _Float16* __restrict__ wtile, int lane, const v16h& B,
    float* __restrict__ m_partial) {
  // --- 16 sequential scan steps; store states (f16) into the LDS tile. ---
#pragma unroll
  for (int r = 0; r < 16; ++r) {
    const int t = t0 + r;               // uniform
    float st = 0.0f;
    if (!GUARD || t < N) {
      const float ht = h[t];            // wave-uniform (scalar) load
      const float up = fmaxf(state, sig2m1((ht - alpha) * TEMP_INV));
      const float dn = fminf(state, sig2m1((ht - beta)  * TEMP_INV));
      state = (ht > field) ? up : dn;
      field = ht;
      st = state;
    }
    wtile[r * TILE_STRIDE + lane] = (_Float16)st;
  }
  // Tile is wave-private: same-wave LDS ops are in-order in HW; this fence only
  // pins compiler ordering (no barrier instruction, waves stay independent).
  __builtin_amdgcn_fence(__ATOMIC_SEQ_CST, "wavefront");

  // --- Gather A per ISA 16-bit A 16x32 layout:
  //     lane l holds row M = l&15; element e -> K = e + 8*(e>=8) + 8*(l>=16).
  v16h A;
  const int m  = lane & 15;
  const int hb = lane >> 4;             // 0 or 1
#pragma unroll
  for (int e = 0; e < 16; ++e) {
    const int kk = e + ((e >= 8) ? 8 : 0) + (hb ? 8 : 0);
    A[e] = wtile[m * TILE_STRIDE + kk];
  }

  // --- D = A x B + 0 : C[m][n] = sum_k state[t0+m][k] * w[k] (same all n). ---
  v8f C = {};
  C = __builtin_amdgcn_wmma_f32_16x16x32_f16(
      /*neg_a=*/false, A, /*neg_b=*/false, B,
      /*c_mod=*/(short)0, C, /*reuse_a=*/false, /*reuse_b=*/false);

  // C layout: VGPR r, lanes 0-15 -> (M=r, N=lane); lanes 16-31 -> (M=r+8).
  // Lane 0 covers times t0..t0+7 (N=0); lane 16 covers t0+8..t0+15.
  if ((lane & 15) == 0) {
    const int tb = t0 + (hb << 3);
#pragma unroll
    for (int r = 0; r < 8; ++r) {
      if (!GUARD || (tb + r) < N) atomicAdd(&m_partial[tb + r], C[r]);
    }
  }
  __builtin_amdgcn_fence(__ATOMIC_SEQ_CST, "wavefront");
}

__global__ void __launch_bounds__(BLOCK_THREADS) preisach_scan_wmma_kernel(
    const float* __restrict__ h, const float* __restrict__ density,
    const float* __restrict__ mesh, float* __restrict__ m_partial,
    int K, int N) {
  // Per-wave 16x32 f16 state tile, row stride 34 halves.
  __shared__ _Float16 tile[WAVES_PER_BLOCK][16 * TILE_STRIDE];

  const int lane = threadIdx.x & 31;
  const int wave = threadIdx.x >> 5;
  const int k    = blockIdx.x * BLOCK_THREADS + threadIdx.x;  // mesh point id
  _Float16* wtile = tile[wave];

  // Per-lane mesh parameters; out-of-range lanes get weight 0 (contribute 0).
  float alpha = 0.0f, beta = 0.0f, w = 0.0f;
  if (k < K) {
    beta  = mesh[2 * k + 0];   // mesh[:,0] = b
    alpha = mesh[2 * k + 1];   // mesh[:,1] = a
    w     = density[k];
  }

  // --- Build B operand: B[kk][n] = w[kk] for all n (replicate over N). ---
  // ISA 32x16 f16 B layout: lane l, element e -> K = e + 8*(e>=8) + 8*(l>=16), N = l&15.
  wtile[lane] = (_Float16)w;           // stage this wave's 32 weights
  __builtin_amdgcn_fence(__ATOMIC_SEQ_CST, "wavefront");
  v16h B;
  {
    const int base = lane & 16;        // 0 for lanes 0-15, 16 for lanes 16-31
#pragma unroll
    for (int e = 0; e < 16; ++e) B[e] = wtile[(e & 7) + ((e >> 3) << 4) + (base >> 1)];
  }
  __builtin_amdgcn_fence(__ATOMIC_SEQ_CST, "wavefront");

  float state = -1.0f;                 // init_state
  float field = 0.0f;                  // init_field (h_{-1})

  // Fast path: full 16-step chunks with no bound checks (all of N=2048).
  const int nfull = N >> 4;
  for (int c = 0; c < nfull; ++c)
    preisach_chunk<false>(c * 16, N, h, alpha, beta, state, field,
                          wtile, lane, B, m_partial);
  // Guarded tail (not taken for N % 16 == 0, kept for generality).
  if (N & 15)
    preisach_chunk<true>(nfull * 16, N, h, alpha, beta, state, field,
                         wtile, lane, B, m_partial);
}

__global__ void __launch_bounds__(256) finalize_kernel(
    const float* __restrict__ m_partial, const float* __restrict__ sum_slot,
    const float* __restrict__ h, float* __restrict__ out, int N) {
  int t = blockIdx.x * blockDim.x + threadIdx.x;
  if (t < N) {
    float m = m_partial[t] / sum_slot[0];
    out[t] = M_SCALE * m + OFFSET_C + h[t] * H_SLOPE;
  }
}

extern "C" void kernel_launch(void* const* d_in, const int* in_sizes, int n_in,
                              void* d_out, int out_size, void* d_ws, size_t ws_size,
                              hipStream_t stream) {
  const float* h       = (const float*)d_in[0];  // (N,1) f32
  const float* density = (const float*)d_in[1];  // (K,)  f32
  const float* mesh    = (const float*)d_in[2];  // (K,2) f32
  const int N = in_sizes[0];                     // 2048
  const int K = in_sizes[1];                     // ~20301

  float* ws        = (float*)d_ws;
  float* m_partial = ws;                         // N floats
  float* sum_slot  = ws + N;                     // 1 float
  float* out       = (float*)d_out;

  // 1) zero per-timestep accumulator + sum slot (ws is poisoned by harness)
  zero_ws_kernel<<<(N + 1 + 255) / 256, 256, 0, stream>>>(ws, N + 1);
  // 2) sum(density)
  density_sum_kernel<<<1, 256, 0, stream>>>(density, sum_slot, K);
  // 3) hysteresis scan + WMMA matvec accumulation
  const int nblocks = (K + BLOCK_THREADS - 1) / BLOCK_THREADS;
  preisach_scan_wmma_kernel<<<nblocks, BLOCK_THREADS, 0, stream>>>(
      h, density, mesh, m_partial, K, N);
  // 4) out[t] = m[t]/sum + h[t]
  finalize_kernel<<<(N + 255) / 256, 256, 0, stream>>>(m_partial, sum_slot, h, out, N);
}
